// DeepSeekSparseAttention_3324304687166
// MI455X (gfx1250) — matrix operand order repaired and checked by reading, not compiled, not run on hardware
//
#include <hip/hip_runtime.h>
#include <stdint.h>

typedef __bf16 bf16;
typedef __attribute__((ext_vector_type(16))) __bf16 v16bf;
typedef __attribute__((ext_vector_type(8)))  float  v8f;
typedef unsigned int u32v4 __attribute__((ext_vector_type(4)));
typedef int i32v4 __attribute__((ext_vector_type(4)));
typedef int i32v8 __attribute__((ext_vector_type(8)));

#if defined(__has_builtin)
#if __has_builtin(__builtin_amdgcn_tensor_load_to_lds)
#define HAVE_TDM 1
#endif
#if __has_builtin(__builtin_amdgcn_s_wait_asynccnt)
#define HAVE_WAIT_ASYNC_BUILTIN 1
#endif
#if __has_builtin(__builtin_amdgcn_s_wait_tensorcnt)
#define HAVE_WAIT_TENSOR_BUILTIN 1
#endif
#endif

namespace {
constexpr int kSeq  = 4096;   // L
constexpr int kD    = 512;    // d_model
constexpr int kH    = 8;      // heads
constexpr int kHd   = 64;     // head dim
constexpr int kNb   = 128;    // 32-token blocks
constexpr int kSel  = 8;      // top-k blocks per query block
}

union V16U { v16bf v; uint4 q[2]; };

// A-matrix (16x32 bf16) fragment: lane row = lane&15;
// lanes<16 hold K[0..15]; lanes>=16 hold K[16..31] (contiguous, matches B).
__device__ __forceinline__ v16bf load_a16x32(const bf16* base, int ld, int lane) {
  V16U u;
  const bf16* p = base + (size_t)(lane & 15) * (size_t)ld;
  const int g = (lane >> 4) & 1;
  u.q[0] = *(const uint4*)(p + 16 * g);
  u.q[1] = *(const uint4*)(p + 16 * g + 8);
  return u.v;
}

// B-matrix (32x16 bf16) fragment from a K-contiguous ("B-transposed") layout
// baseT[n][k]: lane col = lane&15, holds K[16*(lane>>4) .. +15] contiguously.
__device__ __forceinline__ v16bf load_bT32x16(const bf16* baseT, int ld, int lane) {
  V16U u;
  const bf16* p = baseT + (size_t)(lane & 15) * (size_t)ld + 16 * ((lane >> 4) & 1);
  u.q[0] = *(const uint4*)(p + 0);
  u.q[1] = *(const uint4*)(p + 8);
  return u.v;
}

__device__ __forceinline__ v8f wmma_bf16(v16bf a, v16bf b, v8f c) {
  return __builtin_amdgcn_wmma_f32_16x16x32_bf16(false, a, false, b, (short)0, c, false, false);
}

__device__ __forceinline__ float bf_round(float x) { return (float)(bf16)x; }

__device__ __forceinline__ void wait_asynccnt0() {
#if defined(HAVE_WAIT_ASYNC_BUILTIN)
  __builtin_amdgcn_s_wait_asynccnt(0);
#else
  asm volatile("s_wait_asynccnt 0x0" ::: "memory");
#endif
}

__device__ __forceinline__ void wait_tensorcnt0() {
#if defined(HAVE_WAIT_TENSOR_BUILTIN)
  __builtin_amdgcn_s_wait_tensorcnt(0);
#else
  asm volatile("s_wait_tensorcnt 0x0" ::: "memory");
#endif
}

// ---------------- data prep ----------------

__global__ __launch_bounds__(256) void pack_x_kernel(const float* __restrict__ x,
                                                     bf16* __restrict__ xb) {
  int i = blockIdx.x * 256 + threadIdx.x;           // < 4096*512
  xb[i] = (bf16)x[i];
}

__global__ __launch_bounds__(256) void transpose_w_kernel(
    const float* __restrict__ wq, const float* __restrict__ wk,
    const float* __restrict__ wv, const float* __restrict__ wo,
    bf16* __restrict__ wqT, bf16* __restrict__ wkT,
    bf16* __restrict__ wvT, bf16* __restrict__ woT) {
  int i = blockIdx.x * 256 + threadIdx.x;           // i = k*512 + n
  int k = i >> 9, n = i & 511;
  const float* src = (blockIdx.y == 0) ? wq : (blockIdx.y == 1) ? wk
                    : (blockIdx.y == 2) ? wv : wo;
  bf16* dst = (blockIdx.y == 0) ? wqT : (blockIdx.y == 1) ? wkT
             : (blockIdx.y == 2) ? wvT : woT;
  dst[(size_t)n * kD + k] = (bf16)src[i];
}

// ---------------- indexer path (f32, matches reference precision) ----------------

__global__ __launch_bounds__(256) void mean_blocks_kernel(const float* __restrict__ x,
                                                          float* __restrict__ xblk) {
  int i = blockIdx.x * 256 + threadIdx.x;           // < 128*512
  int b = i >> 9, d = i & 511;
  float s = 0.0f;
  for (int t = 0; t < 32; ++t) s += x[(size_t)(b * 32 + t) * kD + d];
  xblk[i] = s * (1.0f / 32.0f);
}

__global__ __launch_bounds__(256) void idx_proj_kernel(
    const float* __restrict__ xblk, const float* __restrict__ iq,
    const float* __restrict__ ik, const float* __restrict__ ig,
    float* __restrict__ qi, float* __restrict__ ki, float* __restrict__ wg) {
  int i = blockIdx.x * 256 + threadIdx.x;
  if (blockIdx.y < 2) {                              // qi / ki: 128x512
    int b = i >> 9, c = i & 511;
    const float* w = blockIdx.y ? ik : iq;
    float s = 0.0f;
    for (int k = 0; k < kD; ++k) s += xblk[(size_t)b * kD + k] * w[(size_t)k * kD + c];
    (blockIdx.y ? ki : qi)[i] = s;
  } else {                                           // wgt: 128x4
    if (i < kNb * 4) {
      int b = i >> 2, c = i & 3;
      float s = 0.0f;
      for (int k = 0; k < kD; ++k) s += xblk[(size_t)b * kD + k] * ig[k * 4 + c];
      wg[i] = s;
    }
  }
}

__global__ __launch_bounds__(128) void topk_kernel(
    const float* __restrict__ qi, const float* __restrict__ ki,
    const float* __restrict__ wg, int* __restrict__ topidx) {
  const int q = blockIdx.x;                          // query block
  const int k = threadIdx.x;                         // key block
  __shared__ float s_q[kD];
  __shared__ float s_w[4];
  __shared__ float s_score[kNb];
  __shared__ float rv[kNb];
  __shared__ int   ri[kNb];
  for (int i = k; i < kD; i += 128) s_q[i] = qi[(size_t)q * kD + i];
  if (k < 4) s_w[k] = wg[q * 4 + k];
  __syncthreads();
  float sc = 0.0f;
#pragma unroll
  for (int hh = 0; hh < 4; ++hh) {
    float d = 0.0f;
    const float* kp = ki + (size_t)k * kD + hh * 128;
    const float* qp = s_q + hh * 128;
    for (int dd = 0; dd < 128; ++dd) d += qp[dd] * kp[dd];
    sc += fmaxf(d, 0.0f) * s_w[hh];
  }
  s_score[k] = sc;
  __syncthreads();
  for (int it = 0; it < kSel; ++it) {
    rv[k] = s_score[k]; ri[k] = k;
    __syncthreads();
    for (int s = 64; s > 0; s >>= 1) {
      if (k < s && rv[k + s] > rv[k]) { rv[k] = rv[k + s]; ri[k] = ri[k + s]; }
      __syncthreads();
    }
    if (k == 0) { topidx[q * kSel + it] = ri[0]; s_score[ri[0]] = -3.0e38f; }
    __syncthreads();
  }
}

// ---------------- QKV projection + RoPE (WMMA bf16) ----------------
// One wave computes a 16(row) x 64(head-dim) tile of Q, K and V for one head.
// qr/kr: [h][pos][d] bf16 (K-contiguous => usable as B^T for Q.K^T)
// vT:    [h][d][pos] bf16 (so P.V's B columns are pos-contiguous)

__global__ __launch_bounds__(32) void qkv_rope_kernel(
    const bf16* __restrict__ xb, const bf16* __restrict__ wqT,
    const bf16* __restrict__ wkT, const bf16* __restrict__ wvT,
    bf16* __restrict__ qr, bf16* __restrict__ kr, bf16* __restrict__ vT) {
  const int lane = threadIdx.x;
  const int m0 = blockIdx.x * 16;
  const int h  = blockIdx.y;
  const int n0 = h * kHd;
  v8f cq[4] = {}, ck[4] = {}, cv[4] = {};
  for (int k0 = 0; k0 < kD; k0 += 32) {
    v16bf a = load_a16x32(xb + (size_t)m0 * kD + k0, kD, lane);
#pragma unroll
    for (int nc = 0; nc < 4; ++nc) {
      const size_t wo_ = (size_t)(n0 + nc * 16) * kD + k0;
      cq[nc] = wmma_bf16(a, load_bT32x16(wqT + wo_, kD, lane), cq[nc]);
      ck[nc] = wmma_bf16(a, load_bT32x16(wkT + wo_, kD, lane), ck[nc]);
      cv[nc] = wmma_bf16(a, load_bT32x16(wvT + wo_, kD, lane), cv[nc]);
    }
  }
  const int g = lane >> 4, nl = lane & 15;
  float invf[2];
  invf[0] = __powf(10000.0f, -((float)nl) * (1.0f / 32.0f));
  invf[1] = __powf(10000.0f, -((float)(nl + 16)) * (1.0f / 32.0f));
#pragma unroll
  for (int r = 0; r < 8; ++r) {
    const int pos = m0 + r + 8 * g;                  // C layout: comp r -> row r / r+8
    const size_t qko = ((size_t)h * kSeq + pos) * kHd;
#pragma unroll
    for (int pr = 0; pr < 2; ++pr) {                 // dim pairs (j, j+32), j = pr*16+nl
      float sn, cs;
      __sincosf((float)pos * invf[pr], &sn, &cs);
      float q1 = bf_round(cq[pr][r]), q2 = bf_round(cq[pr + 2][r]);
      float k1 = bf_round(ck[pr][r]), k2 = bf_round(ck[pr + 2][r]);
      qr[qko + pr * 16 + nl]      = (bf16)(q1 * cs - q2 * sn);
      qr[qko + pr * 16 + nl + 32] = (bf16)(q1 * sn + q2 * cs);
      kr[qko + pr * 16 + nl]      = (bf16)(k1 * cs - k2 * sn);
      kr[qko + pr * 16 + nl + 32] = (bf16)(k1 * sn + k2 * cs);
    }
#pragma unroll
    for (int dc = 0; dc < 4; ++dc)
      vT[((size_t)h * kHd + dc * 16 + nl) * kSeq + pos] = (bf16)cv[dc][r];
  }
}

// ---------------- sparse attention ----------------
// One wave = 16 query rows x 1 head x 256 gathered keys.
// K blocks staged global->LDS with async-load (ASYNCcnt); V^T tiles staged with
// the Tensor Data Mover (TENSORcnt) when the builtin exists.

__global__ __launch_bounds__(32) void attn_kernel(
    const bf16* __restrict__ qr, const bf16* __restrict__ kr,
    const bf16* __restrict__ vT, const int* __restrict__ topidx,
    bf16* __restrict__ ao) {
  __shared__ alignas(16) bf16 sP[16][256];          // P tile (8 KB)
  __shared__ alignas(16) bf16 sK[kSel * 32 * kHd];  // gathered K  [key][d] (32 KB)
  __shared__ alignas(16) bf16 sV[kSel * kHd * 32];  // gathered V^T [sb][d][pos] (32 KB)
  const int lane = threadIdx.x;
  const int m0 = blockIdx.x * 16;
  const int h  = blockIdx.y;
  const int qb = m0 >> 5;                            // 32-token query block
  int sel[kSel];
#pragma unroll
  for (int j = 0; j < kSel; ++j) sel[j] = topidx[qb * kSel + j];

  // ---- stage gathered V^T tiles via TDM (64 lines x 32 contiguous elems each)
#pragma unroll
  for (int sb = 0; sb < kSel; ++sb) {
    const int key0 = sel[sb] * 32;
#if defined(HAVE_TDM)
    const uint64_t ga = (uint64_t)(uintptr_t)(vT + ((size_t)h * kHd) * kSeq + key0);
    const uint32_t la = (uint32_t)(uintptr_t)(&sV[sb * kHd * 32]);
    u32v4 g0;
    g0[0] = 1u;                                          // count=1, no gather/iterate
    g0[1] = la;                                          // lds_addr
    g0[2] = (uint32_t)ga;                                // global_addr[31:0]
    g0[3] = (uint32_t)((ga >> 32) & 0x01FFFFFFu) | (2u << 30);  // addr[56:32] | type=2
    i32v8 g1;
    g1[0] = (int)(1u << 16);                             // wg_mask=0, data_size=2B
    g1[1] = (int)((kSeq & 0xFFFF) << 16);                // tensor_dim0[15:0]
    g1[2] = (int)((kSeq >> 16) | ((kH * kHd & 0xFFFF) << 16)); // t0[31:16]|t1[15:0]
    g1[3] = (int)(32u << 16);                            // t1[31:16]=0 | tile_dim0=32
    g1[4] = (int)(kHd);                                  // tile_dim1=64, tile_dim2=0
    g1[5] = (int)(kSeq);                                 // tensor_dim0_stride[31:0]
    g1[6] = 0;                                           // stride0[47:32]|stride1[15:0]
    g1[7] = 0;
    i32v4 g2 = {0, 0, 0, 0};
    i32v4 g3 = {0, 0, 0, 0};
#if __clang_major__ >= 23
    i32v8 g4 = {0, 0, 0, 0, 0, 0, 0, 0};
    __builtin_amdgcn_tensor_load_to_lds(g0, g1, g2, g3, g4, 0);
#else
    __builtin_amdgcn_tensor_load_to_lds(g0, g1, g2, g3, 0);
#endif
#else
    const bf16* gv = vT + ((size_t)h * kHd) * kSeq + key0;
    for (int i = lane; i < kHd * 32; i += 32)
      sV[sb * kHd * 32 + i] = gv[(size_t)(i >> 5) * kSeq + (i & 31)];
#endif
  }

  // ---- stage gathered K blocks via async global->LDS (b128 per lane)
#pragma unroll
  for (int sb = 0; sb < kSel; ++sb) {
    const uint64_t ga =
        (uint64_t)(uintptr_t)(kr + ((size_t)h * kSeq + sel[sb] * 32) * kHd);
    const uint32_t la = (uint32_t)(uintptr_t)(&sK[sb * 32 * kHd]);
#pragma unroll
    for (int it = 0; it < 8; ++it) {                 // 8 x 512B = one 32x64 bf16 block
      uint32_t ldst = la + (uint32_t)(it * 512 + lane * 16);
      uint64_t gsrc = ga + (uint64_t)(it * 512 + lane * 16);
      asm volatile("global_load_async_to_lds_b128 %0, %1, off"
                   :: "v"(ldst), "v"(gsrc) : "memory");
    }
  }
  wait_asynccnt0();
  __syncthreads();

  const bf16* qbase = qr + ((size_t)h * kSeq + m0) * kHd;
  v16bf aq[2];
  aq[0] = load_a16x32(qbase, kHd, lane);
  aq[1] = load_a16x32(qbase + 32, kHd, lane);

  // scores: 8 selected blocks x (2 tiles of 16 keys), K fragments from LDS
  v8f S[16] = {};
#pragma unroll
  for (int sb = 0; sb < kSel; ++sb) {
#pragma unroll
    for (int nc = 0; nc < 2; ++nc) {
      v8f acc = {};
#pragma unroll
      for (int kc = 0; kc < 2; ++kc) {
        v16bf b = load_bT32x16(&sK[(sb * 32 + nc * 16) * kHd] + kc * 32, kHd, lane);
        acc = wmma_bf16(aq[kc], b, acc);
      }
#pragma unroll
      for (int r = 0; r < 8; ++r) acc[r] *= 0.125f;  // 1/sqrt(64)
      S[sb * 2 + nc] = acc;
    }
  }

  // softmax over 256 keys; rows live in comp r of each 16-lane half
  float mx[8], sm[8];
#pragma unroll
  for (int r = 0; r < 8; ++r) { mx[r] = -3.0e38f; sm[r] = 0.0f; }
#pragma unroll
  for (int t = 0; t < 16; ++t)
#pragma unroll
    for (int r = 0; r < 8; ++r) mx[r] = fmaxf(mx[r], S[t][r]);
#pragma unroll
  for (int off = 8; off >= 1; off >>= 1)
#pragma unroll
    for (int r = 0; r < 8; ++r) mx[r] = fmaxf(mx[r], __shfl_xor(mx[r], off, 16));
#pragma unroll
  for (int t = 0; t < 16; ++t)
#pragma unroll
    for (int r = 0; r < 8; ++r) {
      float e = __expf(S[t][r] - mx[r]); S[t][r] = e; sm[r] += e;
    }
#pragma unroll
  for (int off = 8; off >= 1; off >>= 1)
#pragma unroll
    for (int r = 0; r < 8; ++r) sm[r] += __shfl_xor(sm[r], off, 16);

  const int g = lane >> 4, nl = lane & 15;
#pragma unroll
  for (int t = 0; t < 16; ++t)                       // C layout -> LDS row-major
#pragma unroll
    for (int r = 0; r < 8; ++r) sP[r + 8 * g][t * 16 + nl] = (bf16)S[t][r];
  __syncthreads();
  wait_tensorcnt0();                                 // V^T tiles resident in LDS

  // O = P.V  (A = P from LDS, B = staged V^T: [d][pos] contiguous, ld=32)
  v8f O[4] = {};
#pragma unroll
  for (int sb = 0; sb < kSel; ++sb) {
    v16bf p = load_a16x32(&sP[0][0] + sb * 32, 256, lane);
#pragma unroll
    for (int dc = 0; dc < 4; ++dc) {
      v16bf b = load_bT32x16(&sV[(sb * kHd + dc * 16) * 32], 32, lane);
      O[dc] = wmma_bf16(p, b, O[dc]);
    }
  }
#pragma unroll
  for (int r = 0; r < 8; ++r) {
    const float rinv = 1.0f / sm[r];
    const int pos = m0 + r + 8 * g;
#pragma unroll
    for (int dc = 0; dc < 4; ++dc)
      ao[(size_t)pos * kD + h * kHd + dc * 16 + nl] = (bf16)(O[dc][r] * rinv);
  }
}

// ---------------- output projection (WMMA bf16) ----------------

__global__ __launch_bounds__(32) void out_proj_kernel(
    const bf16* __restrict__ ao, const bf16* __restrict__ woT,
    bf16* __restrict__ out) {
  const int lane = threadIdx.x;
  const int m0 = blockIdx.x * 16;
  const int n0 = blockIdx.y * 64;
  v8f c[4] = {};
  for (int k0 = 0; k0 < kD; k0 += 32) {
    __builtin_prefetch(ao + (size_t)m0 * kD + k0 + 64, 0, 1);  // global_prefetch_b8
    v16bf a = load_a16x32(ao + (size_t)m0 * kD + k0, kD, lane);
#pragma unroll
    for (int nc = 0; nc < 4; ++nc)
      c[nc] = wmma_bf16(a, load_bT32x16(woT + (size_t)(n0 + nc * 16) * kD + k0,
                                        kD, lane), c[nc]);
  }
  const int g = lane >> 4, nl = lane & 15;
#pragma unroll
  for (int r = 0; r < 8; ++r) {
    const int pos = m0 + r + 8 * g;
#pragma unroll
    for (int nc = 0; nc < 4; ++nc)
      out[(size_t)pos * kD + n0 + nc * 16 + nl] = (bf16)c[nc][r];
  }
}

// ---------------- launcher ----------------

extern "C" void kernel_launch(void* const* d_in, const int* in_sizes, int n_in,
                              void* d_out, int out_size, void* d_ws, size_t ws_size,
                              hipStream_t stream) {
  (void)in_sizes; (void)n_in; (void)out_size; (void)ws_size;
  const float* x  = (const float*)d_in[0];
  const float* wq = (const float*)d_in[1];
  const float* wk = (const float*)d_in[2];
  const float* wv = (const float*)d_in[3];
  const float* wo = (const float*)d_in[4];
  const float* iq = (const float*)d_in[5];
  const float* ik = (const float*)d_in[6];
  const float* ig = (const float*)d_in[7];

  uint8_t* base = (uint8_t*)d_ws;
  size_t off = 0;
  auto carve = [&](size_t bytes) -> uint8_t* {
    uint8_t* p = base + off;
    off = (off + bytes + 255) & ~(size_t)255;
    return p;
  };
  bf16* xb   = (bf16*)carve((size_t)kSeq * kD * 2);        // x in bf16
  bf16* wqT  = (bf16*)carve((size_t)kD * kD * 2);
  bf16* wkT  = (bf16*)carve((size_t)kD * kD * 2);
  bf16* wvT  = (bf16*)carve((size_t)kD * kD * 2);
  bf16* woT  = (bf16*)carve((size_t)kD * kD * 2);
  bf16* qrp  = (bf16*)carve((size_t)kH * kSeq * kHd * 2);  // roped Q [h][pos][d]
  bf16* krp  = (bf16*)carve((size_t)kH * kSeq * kHd * 2);  // roped K [h][pos][d]
  bf16* vT   = (bf16*)carve((size_t)kH * kHd * kSeq * 2);  // V^T    [h][d][pos]
  bf16* ao   = (bf16*)carve((size_t)kSeq * kD * 2);        // attention out
  float* xblk = (float*)carve((size_t)kNb * kD * 4);
  float* qi   = (float*)carve((size_t)kNb * kD * 4);
  float* ki   = (float*)carve((size_t)kNb * kD * 4);
  float* wg   = (float*)carve((size_t)kNb * 4 * 4);
  int*   top  = (int*)carve((size_t)kNb * kSel * 4);

  pack_x_kernel<<<(kSeq * kD) / 256, 256, 0, stream>>>(x, xb);
  transpose_w_kernel<<<dim3((kD * kD) / 256, 4), 256, 0, stream>>>(
      wq, wk, wv, wo, wqT, wkT, wvT, woT);

  mean_blocks_kernel<<<(kNb * kD) / 256, 256, 0, stream>>>(x, xblk);
  idx_proj_kernel<<<dim3((kNb * kD) / 256, 3), 256, 0, stream>>>(
      xblk, iq, ik, ig, qi, ki, wg);
  topk_kernel<<<kNb, 128, 0, stream>>>(qi, ki, wg, top);

  qkv_rope_kernel<<<dim3(kSeq / 16, kH), 32, 0, stream>>>(
      xb, wqT, wkT, wvT, qrp, krp, vT);
  attn_kernel<<<dim3(kSeq / 16, kH), 32, 0, stream>>>(qrp, krp, vT, top, ao);
  out_proj_kernel<<<dim3(kSeq / 16, kD / 64), 32, 0, stream>>>(
      ao, woT, (bf16*)d_out);
}
